// GCN_59459527246262
// MI455X (gfx1250) — compile-verified, matrix operand order checked by reading, not run on hardware
//
#include <hip/hip_runtime.h>
#include <hip/hip_bf16.h>

// ---------------------------------------------------------------------------
// 2-layer GCN for MI455X (gfx1250, wave32, WMMA).
//   N=100000 nodes, E=1.6M edges, F_IN=256, H=128, C_OUT=2
// Pipeline:
//   deg/dinv -> pack W1/W2 into WMMA B-fragments -> WMMA GEMM1 (bf16 in, f32
//   acc) -> self-loop init -> edge scatter (fp32 HW atomics) -> bias+ReLU ->
//   WMMA GEMM2 -> self-loop init -> edge scatter -> bias+softmax.
// ---------------------------------------------------------------------------

typedef __attribute__((ext_vector_type(16))) __bf16 v16bf;
typedef __attribute__((ext_vector_type(8)))  float  v8f;

#define GCN_FIN 256
#define GCN_H   128
#define GCN_CO  2

// ---------------------------------------------------------------- degree ----
__global__ void k_deg_init(float* deg, int n) {
    int i = blockIdx.x * blockDim.x + threadIdx.x;
    if (i < n) deg[i] = 1.0f;  // self loop
}

__global__ void k_deg_accum(const long long* __restrict__ ei, float* deg,
                            int n, long long E) {
    long long e = (long long)blockIdx.x * blockDim.x + threadIdx.x;
    if (e >= E) return;
    long long d = ei[E + e];  // dst row
    unsafeAtomicAdd(&deg[d], 1.0f);
}

__global__ void k_dinv(const float* __restrict__ deg, float* dinv, int n) {
    int i = blockIdx.x * blockDim.x + threadIdx.x;
    if (i < n) dinv[i] = rsqrtf(deg[i]);  // deg >= 1 always (self loop)
}

// ------------------------------------------------- pack W into B-fragments ---
// Fragment layout for V_WMMA_F32_16X16X32_BF16 B operand (32x16, bf16):
//   lane: n = ct*16 + (lane&15); kbase = ks*32 + 8*(lane>=16)
//   elems 0..7  -> K = kbase+e ;  elems 8..15 -> K = kbase+16+(e-8)
// Stored as frag[((ct*ksteps)+ks)*32 + lane][16] (32B per lane chunk).
__global__ void k_pack_bfrag(const float* __restrict__ W, __bf16* __restrict__ frag,
                             int K, int ncols, int nColTiles) {
    int ksteps = K >> 5;
    int total  = nColTiles * ksteps * 32;
    int idx = blockIdx.x * blockDim.x + threadIdx.x;
    if (idx >= total) return;
    int lane = idx & 31;
    int t    = idx >> 5;
    int ks   = t % ksteps;
    int ct   = t / ksteps;
    int n    = ct * 16 + (lane & 15);
    int kb   = ks * 32 + ((lane >> 4) << 3);
    __bf16* out = frag + (size_t)idx * 16;
#pragma unroll
    for (int e = 0; e < 8; ++e)
        out[e]     = (n < ncols) ? (__bf16)W[(size_t)(kb + e) * ncols + n] : (__bf16)0.0f;
#pragma unroll
    for (int e = 0; e < 8; ++e)
        out[8 + e] = (n < ncols) ? (__bf16)W[(size_t)(kb + 16 + e) * ncols + n] : (__bf16)0.0f;
}

// ------------------------------------------------------------ WMMA GEMM ----
// One wave computes a 16-row stripe across NCT 16-col tiles: A rows are read
// from HBM exactly once, B fragments are tiny and L0-resident.
template <int NCT>
__global__ __launch_bounds__(256) void k_wmma_gemm(
    const float* __restrict__ A,       // nRows x K, row major, fp32
    const __bf16* __restrict__ Bfrag,  // packed fragments
    float* __restrict__ C,             // nRows x outCols, row major
    int nRowTiles, int K, int outCols) {
    int wave = threadIdx.x >> 5;
    int lane = threadIdx.x & 31;
    int rowTile = blockIdx.x * 8 + wave;
    if (rowTile >= nRowTiles) return;  // wave-uniform exit (EXEC all-1 for WMMA)
    int rowBase = rowTile * 16;
    int ksteps  = K >> 5;

    v8f acc[NCT];
#pragma unroll
    for (int ct = 0; ct < NCT; ++ct) acc[ct] = (v8f){};

    // A fragment addressing: lane holds row (lane&15), K-halves per hi-bit.
    int row = rowBase + (lane & 15);
    const float* arow = A + (size_t)row * K + ((lane >> 4) << 3);

    for (int ks = 0; ks < ksteps; ++ks) {
        const float* ap = arow + ks * 32;
        v16bf a;
#pragma unroll
        for (int e = 0; e < 8; ++e) a[e]     = (__bf16)ap[e];
#pragma unroll
        for (int e = 0; e < 8; ++e) a[8 + e] = (__bf16)ap[16 + e];
#pragma unroll
        for (int ct = 0; ct < NCT; ++ct) {
            v16bf b = *(const v16bf*)(Bfrag + ((size_t)(ct * ksteps + ks) * 32 + lane) * 16);
            acc[ct] = __builtin_amdgcn_wmma_f32_16x16x32_bf16(
                false, a, false, b, (short)0, acc[ct], false, false);
        }
    }

    // D layout: VGPR r, lane -> (M = rowBase + r + 8*(lane>=16), N = lane&15)
    int nloc = lane & 15;
    int m0   = rowBase + ((lane >> 4) << 3);
#pragma unroll
    for (int ct = 0; ct < NCT; ++ct) {
        int col = ct * 16 + nloc;
        if (col < outCols) {
#pragma unroll
            for (int r = 0; r < 8; ++r)
                C[(size_t)(m0 + r) * outCols + col] = acc[ct][r];
        }
    }
}

// --------------------------------------------- aggregation (layer 1, H=128) -
__global__ void k_self_init128(const float* __restrict__ h,
                               const float* __restrict__ dinv,
                               float* __restrict__ agg, int n) {
    long long idx = (long long)blockIdx.x * blockDim.x + threadIdx.x;
    long long v = idx >> 5;
    if (v >= n) return;
    int cg = (int)(idx & 31) << 2;
    float w = dinv[v] * dinv[v];
    float4 hv = *(const float4*)(h + (size_t)v * GCN_H + cg);
    float4 o = make_float4(w * hv.x, w * hv.y, w * hv.z, w * hv.w);
    *(float4*)(agg + (size_t)v * GCN_H + cg) = o;
}

__global__ void k_scatter128(const long long* __restrict__ ei,
                             const float* __restrict__ h,
                             const float* __restrict__ dinv,
                             float* __restrict__ agg, long long E) {
    long long idx = (long long)blockIdx.x * blockDim.x + threadIdx.x;
    long long e = idx >> 5;
    if (e >= E) return;
    int cg = (int)(idx & 31) << 2;
    long long s = ei[e];
    long long d = ei[E + e];
    float w = dinv[s] * dinv[d];
    float4 hv = *(const float4*)(h + (size_t)s * GCN_H + cg);
    float* ag = agg + (size_t)d * GCN_H + cg;
    unsafeAtomicAdd(ag + 0, w * hv.x);
    unsafeAtomicAdd(ag + 1, w * hv.y);
    unsafeAtomicAdd(ag + 2, w * hv.z);
    unsafeAtomicAdd(ag + 3, w * hv.w);
}

__global__ void k_bias_relu(float* __restrict__ agg, const float* __restrict__ b,
                            long long total) {
    long long i = (long long)blockIdx.x * blockDim.x + threadIdx.x;
    if (i >= total) return;
    float v = agg[i] + b[(int)(i & (GCN_H - 1))];
    agg[i] = v > 0.0f ? v : 0.0f;
}

// --------------------------------------------- aggregation (layer 2, C=2) ---
__global__ void k_self_init2(const float* __restrict__ h2,
                             const float* __restrict__ dinv,
                             float* __restrict__ agg2, int n) {
    int v = blockIdx.x * blockDim.x + threadIdx.x;
    if (v >= n) return;
    float w = dinv[v] * dinv[v];
    agg2[2 * (size_t)v + 0] = w * h2[2 * (size_t)v + 0];
    agg2[2 * (size_t)v + 1] = w * h2[2 * (size_t)v + 1];
}

__global__ void k_scatter2(const long long* __restrict__ ei,
                           const float* __restrict__ h2,
                           const float* __restrict__ dinv,
                           float* __restrict__ agg2, long long E) {
    long long e = (long long)blockIdx.x * blockDim.x + threadIdx.x;
    if (e >= E) return;
    long long s = ei[e];
    long long d = ei[E + e];
    float w = dinv[s] * dinv[d];
    unsafeAtomicAdd(&agg2[2 * (size_t)d + 0], w * h2[2 * (size_t)s + 0]);
    unsafeAtomicAdd(&agg2[2 * (size_t)d + 1], w * h2[2 * (size_t)s + 1]);
}

__global__ void k_bias_softmax2(const float* __restrict__ agg2,
                                const float* __restrict__ b,
                                float* __restrict__ out, int n) {
    int v = blockIdx.x * blockDim.x + threadIdx.x;
    if (v >= n) return;
    float a0 = agg2[2 * (size_t)v + 0] + b[0];
    float a1 = agg2[2 * (size_t)v + 1] + b[1];
    float m = fmaxf(a0, a1);
    float e0 = __expf(a0 - m);
    float e1 = __expf(a1 - m);
    float inv = 1.0f / (e0 + e1);
    out[2 * (size_t)v + 0] = e0 * inv;
    out[2 * (size_t)v + 1] = e1 * inv;
}

// ---------------------------------------------------------------- launch ----
static inline size_t alignUp(size_t x, size_t a) { return (x + a - 1) & ~(a - 1); }

extern "C" void kernel_launch(void* const* d_in, const int* in_sizes, int n_in,
                              void* d_out, int out_size, void* d_ws, size_t ws_size,
                              hipStream_t stream) {
    const float*     x  = (const float*)d_in[0];
    const long long* ei = (const long long*)d_in[1];  // int64 [2, E]
    const float*     W1 = (const float*)d_in[2];
    const float*     b1 = (const float*)d_in[3];
    const float*     W2 = (const float*)d_in[4];
    const float*     b2 = (const float*)d_in[5];
    float* out = (float*)d_out;

    const int       N = in_sizes[0] / GCN_FIN;   // 100000
    const long long E = in_sizes[1] / 2;         // 1600000

    // workspace carve-out (~105 MB)
    char* base = (char*)d_ws;
    size_t off = 0;
    float* deg  = (float*)(base + off); off = alignUp(off + (size_t)N * 4, 256);
    float* dinv = (float*)(base + off); off = alignUp(off + (size_t)N * 4, 256);
    float* h    = (float*)(base + off); off = alignUp(off + (size_t)N * GCN_H * 4, 256);
    float* agg1 = (float*)(base + off); off = alignUp(off + (size_t)N * GCN_H * 4, 256);
    float* h2   = (float*)(base + off); off = alignUp(off + (size_t)N * GCN_CO * 4, 256);
    float* agg2 = (float*)(base + off); off = alignUp(off + (size_t)N * GCN_CO * 4, 256);
    __bf16* fragW1 = (__bf16*)(base + off); off = alignUp(off + (size_t)8 * 8 * 32 * 16 * 2, 256);
    __bf16* fragW2 = (__bf16*)(base + off); off = alignUp(off + (size_t)1 * 4 * 32 * 16 * 2, 256);
    (void)ws_size;

    const int T = 256;
    const int nRowTiles = N / 16;  // 100000/16 = 6250 exact

    // 1) degrees + symmetric norm
    k_deg_init<<<(N + T - 1) / T, T, 0, stream>>>(deg, N);
    k_deg_accum<<<(int)((E + T - 1) / T), T, 0, stream>>>(ei, deg, N, E);
    k_dinv<<<(N + T - 1) / T, T, 0, stream>>>(deg, dinv, N);

    // 2) pack weights into WMMA B fragments
    k_pack_bfrag<<<(8 * 8 * 32 + T - 1) / T, T, 0, stream>>>(W1, fragW1, GCN_FIN, GCN_H, 8);
    k_pack_bfrag<<<(1 * 4 * 32 + T - 1) / T, T, 0, stream>>>(W2, fragW2, GCN_H, GCN_CO, 1);

    // 3) layer 1: h = x @ W1 (WMMA), then norm-aggregate, bias+ReLU
    k_wmma_gemm<8><<<(nRowTiles + 7) / 8, T, 0, stream>>>(x, fragW1, h, nRowTiles, GCN_FIN, GCN_H);
    k_self_init128<<<(int)(((long long)N * 32 + T - 1) / T), T, 0, stream>>>(h, dinv, agg1, N);
    k_scatter128<<<(int)((E * 32 + T - 1) / T), T, 0, stream>>>(ei, h, dinv, agg1, E);
    k_bias_relu<<<(int)(((long long)N * GCN_H + T - 1) / T), T, 0, stream>>>(agg1, b1, (long long)N * GCN_H);

    // 4) layer 2: h2 = h1 @ W2 (WMMA), aggregate, bias+softmax
    k_wmma_gemm<1><<<(nRowTiles + 7) / 8, T, 0, stream>>>(agg1, fragW2, h2, nRowTiles, GCN_H, GCN_CO);
    k_self_init2<<<(N + T - 1) / T, T, 0, stream>>>(h2, dinv, agg2, N);
    k_scatter2<<<(int)((E + T - 1) / T), T, 0, stream>>>(ei, h2, dinv, agg2, E);
    k_bias_softmax2<<<(N + T - 1) / T, T, 0, stream>>>(agg2, b2, out, N);
}
